// SSMBlock_70291434766725
// MI455X (gfx1250) — matrix operand order repaired and checked by reading, not compiled, not run on hardware
//
#include <hip/hip_runtime.h>

typedef __attribute__((ext_vector_type(16))) __bf16 v16bf;
typedef __attribute__((ext_vector_type(8)))  __bf16 v8bf;
typedef __attribute__((ext_vector_type(8)))  float  v8f;
typedef __attribute__((ext_vector_type(4)))  float  v4f;

#define D_DIM 256
#define N_PATCH 4096
#define B_BATCH 32

// ---------------------------------------------------------------------------
// prep: dA = exp(delta*A) (elementwise), dB = delta*B.  f32 + bf16 copies.
// ---------------------------------------------------------------------------
__global__ void ssm_prep_kernel(const float* __restrict__ A,
                                const float* __restrict__ B,
                                const float* __restrict__ delta,
                                float* __restrict__ dAf,
                                __bf16* __restrict__ dAbf,
                                float* __restrict__ dBv) {
    int id = blockIdx.x * 256 + threadIdx.x;          // 65536 total
    float dl = delta[0];
    float v = __expf(dl * A[id]);
    dAf[id]  = v;
    dAbf[id] = (__bf16)v;
    if (id < D_DIM) dBv[id] = dl * B[id];
}

// ---------------------------------------------------------------------------
// proj: flat[row] = dot(in[row,:], pw)/256 + mean(pb).  Wave-per-row, lane
// loads 32B (2x v4f, non-temporal: input is read exactly once) -> fully
// coalesced 1KB per wave.  f32 + bf16 out.
// ---------------------------------------------------------------------------
__global__ void ssm_proj_kernel(const float* __restrict__ in,
                                const float* __restrict__ pw,
                                const float* __restrict__ pb,
                                float* __restrict__ flat,
                                __bf16* __restrict__ flatbf) {
    __shared__ float sb[256];
    int tid = threadIdx.x;
    sb[tid] = pb[tid];
    __syncthreads();
    for (int off = 128; off > 0; off >>= 1) {
        if (tid < off) sb[tid] += sb[tid + off];
        __syncthreads();
    }
    float bmean = sb[0] * (1.0f / 256.0f);

    int lane = tid & 31;
    int row  = blockIdx.x * 8 + (tid >> 5);           // 131072 rows total
    const v4f* p = (const v4f*)(in + (size_t)row * D_DIM + lane * 8);
    const v4f* w = (const v4f*)(pw + lane * 8);
    v4f x0 = __builtin_nontemporal_load(p);
    v4f x1 = __builtin_nontemporal_load(p + 1);
    v4f w0 = w[0], w1 = w[1];
    float acc = x0.x*w0.x + x0.y*w0.y + x0.z*w0.z + x0.w*w0.w
              + x1.x*w1.x + x1.y*w1.y + x1.z*w1.z + x1.w*w1.w;
    for (int off = 16; off > 0; off >>= 1)
        acc += __shfl_xor(acc, off, 32);
    if (lane == 0) {
        float f = acc * (1.0f / 256.0f) + bmean;
        flat[row]   = f;
        flatbf[row] = (__bf16)f;
    }
}

// ---------------------------------------------------------------------------
// gemm256: C(256x256,f32+bf16) = A(bf16) @ B(bf16) via WMMA 16x16x32 bf16.
// One 16x16 tile per wave; 8 waves/block, 32 blocks = 256 tiles.
// A fragment = two 16-B-aligned b128 loads.  Layouts per ISA 7.12.2 (wave32).
// ---------------------------------------------------------------------------
__global__ void ssm_gemm256_kernel(const __bf16* __restrict__ Ab,
                                   const __bf16* __restrict__ Bb,
                                   float* __restrict__ Cf,
                                   __bf16* __restrict__ Cb) {
    int tid  = threadIdx.x;
    int lane = tid & 31;
    int wv   = blockIdx.x * 8 + (tid >> 5);           // 0..255
    int mb = wv >> 4, nb = wv & 15;
    int m = lane & 15, half = lane >> 4;
    const __bf16* arow = Ab + (size_t)(mb * 16 + m) * D_DIM;
    int ncol = nb * 16 + m;
    v8f acc = {};
    for (int k0 = 0; k0 < D_DIM; k0 += 32) {
        int kb = k0 + half * 8;
        v8bf lo = *(const v8bf*)(arow + kb);          // K = kb..kb+7   (16B aligned)
        v8bf hi = *(const v8bf*)(arow + kb + 16);     // K = kb+16..kb+23
        v16bf a, b;
#pragma unroll
        for (int i = 0; i < 8; ++i) { a[i] = lo[i]; a[8 + i] = hi[i]; }
        int kf = k0 + half * 8;
#pragma unroll
        for (int j = 0; j < 16; ++j) b[j] = Bb[(size_t)(kf + j + (j >> 3) * 8) * D_DIM + ncol];
        acc = __builtin_amdgcn_wmma_f32_16x16x32_bf16(false, a, false, b,
                                                      (short)0, acc, false, false);
    }
#pragma unroll
    for (int v = 0; v < 8; ++v) {
        int r = mb * 16 + v + 8 * half;
        float val = acc[v];
        Cf[(size_t)r * D_DIM + ncol] = val;
        Cb[(size_t)r * D_DIM + ncol] = (__bf16)val;
    }
}

// ---------------------------------------------------------------------------
// chains: both blocks run 64 steps of  vec <- vec @ M  with M staged in LDS
// as bf16 (128 KB).  block0: M = P = dA^64 (row-major), s_0 = C -> Smat.
// block1: M = dA^T (transposed during preload), v_0 = dB -> Vmat
// (v' = dA v  ==  v' = v @ dA^T).  Thread map j=tid&255, p=tid>>8: a wave's
// 32 lanes read 16 consecutive LDS dwords (pairs share a dword -> broadcast).
// ---------------------------------------------------------------------------
extern __shared__ char chains_smem[];
__global__ void ssm_chains_kernel(const __bf16* __restrict__ Pbf,
                                  const __bf16* __restrict__ dAbf,
                                  const float* __restrict__ Cv,
                                  const float* __restrict__ dBv,
                                  float* __restrict__ Smat,
                                  float* __restrict__ Vmat) {
    __bf16* Msh = (__bf16*)chains_smem;                 // 65536 elems = 128 KB
    float*  vec = (float*)(chains_smem + 131072);       // 256
    float*  part = vec + 256;                           // 1024
    int tid = threadIdx.x;

    if (blockIdx.x == 0) {
        for (int idx = tid; idx < 65536; idx += 1024) Msh[idx] = Pbf[idx];
    } else {
        for (int idx = tid; idx < 65536; idx += 1024) {
            int i = idx >> 8, j = idx & 255;
            Msh[j * 256 + i] = dAbf[idx];               // store transposed
        }
    }
    const float* v0   = (blockIdx.x == 0) ? Cv   : dBv;
    float*       Omat = (blockIdx.x == 0) ? Smat : Vmat;
    if (tid < 256) { vec[tid] = v0[tid]; Omat[tid] = v0[tid]; }
    __syncthreads();

    int j = tid & 255, p = tid >> 8;                    // 4-way split-K
    for (int q = 1; q < 64; ++q) {
        float acc = 0.f;
#pragma unroll 8
        for (int ii = 0; ii < 64; ++ii) {
            int i = p * 64 + ii;
            acc = fmaf(vec[i], (float)Msh[i * 256 + j], acc);
        }
        part[p * 256 + j] = acc;
        __syncthreads();
        if (tid < 256) {
            float s = part[tid] + part[256 + tid] + part[512 + tid] + part[768 + tid];
            vec[tid] = s;
            Omat[q * 256 + tid] = s;
        }
        __syncthreads();
    }
}

// ---------------------------------------------------------------------------
// outer: kernel[64q+r] = s_q . v_r
// ---------------------------------------------------------------------------
__global__ void ssm_outer_kernel(const float* __restrict__ Smat,
                                 const float* __restrict__ Vmat,
                                 float* __restrict__ kern) {
    int t = blockIdx.x * blockDim.x + threadIdx.x;    // 4096
    const float* s = Smat + (t >> 6) * 256;
    const float* v = Vmat + (t & 63) * 256;
    float acc = 0.f;
#pragma unroll 8
    for (int i = 0; i < 256; ++i) acc = fmaf(s[i], v[i], acc);
    kern[t] = acc;
}

// ---------------------------------------------------------------------------
// mkgpad: fold fwd+bwd kernels: g[d]=0.5*(k[d]+k[4095-d]); zero-pad d<0 region.
// gpad[i] = (i<4096) ? 0 : g[i-4096]
// ---------------------------------------------------------------------------
__global__ void ssm_mkgpad_kernel(const float* __restrict__ kern,
                                  __bf16* __restrict__ gpad) {
    int idx = blockIdx.x * blockDim.x + threadIdx.x;  // 8192
    float v = 0.f;
    if (idx >= 4096) v = 0.5f * (kern[idx - 4096] + kern[8191 - idx]);
    gpad[idx] = (__bf16)v;
}

// ---------------------------------------------------------------------------
// conv: res(32 x 4096) = Flat(32 x 4096) @ Toeplitz(g), causal, via WMMA.
// One 16x16 output tile per wave (512 tiles); A fragment = two b128 loads,
// B-tile gathered per-lane from the 16 KB LDS copy of gpad.  The k-loop
// covers only t0 <= n0+15 (causal); boundary zeros come from the pad region.
// ---------------------------------------------------------------------------
__global__ void ssm_conv_kernel(const __bf16* __restrict__ flatbf,
                                const __bf16* __restrict__ gpad,
                                float* __restrict__ res) {
    __shared__ __bf16 gsh[8192];
    int tid = threadIdx.x;
    for (int i = tid; i < 8192; i += 256) gsh[i] = gpad[i];
    __syncthreads();

    int lane = tid & 31, half = lane >> 4, m = lane & 15;
    int wv = blockIdx.x * 8 + (tid >> 5);             // 0..511
    int b0 = (wv & 1) * 16;
    int n0 = (wv >> 1) * 16;
    const __bf16* arow = flatbf + (size_t)(b0 + m) * N_PATCH;
    int ncol = n0 + m;
    v8f acc = {};
    int ksteps = (n0 + 15) / 32 + 1;
    for (int kk = 0; kk < ksteps; ++kk) {
        int t0 = kk * 32;
        int kb = t0 + half * 8;
        v8bf lo = *(const v8bf*)(arow + kb);          // 16B aligned
        v8bf hi = *(const v8bf*)(arow + kb + 16);
        v16bf a, b;
#pragma unroll
        for (int i = 0; i < 8; ++i) { a[i] = lo[i]; a[8 + i] = hi[i]; }
        int base = 4096 + ncol - t0 - half * 8;       // gpad index for j=0
#pragma unroll
        for (int j = 0; j < 16; ++j) b[j] = gsh[base - j - (j >> 3) * 8];
        acc = __builtin_amdgcn_wmma_f32_16x16x32_bf16(false, a, false, b,
                                                      (short)0, acc, false, false);
    }
#pragma unroll
    for (int v = 0; v < 8; ++v)
        res[(size_t)(b0 + v + 8 * half) * N_PATCH + n0 + m] = acc[v];
}

// ---------------------------------------------------------------------------
// bcast: out[b,n,d] = res[b,n] for all d.  128 MiB of non-temporal 16-B
// stores (write-once stream; keep it out of L2's way).
// ---------------------------------------------------------------------------
__global__ void ssm_bcast_kernel(const float* __restrict__ res,
                                 v4f* __restrict__ out) {
    size_t idx = (size_t)blockIdx.x * blockDim.x + threadIdx.x;  // 8388608 v4f
    float v = res[idx >> 6];                                     // 64 v4f per row
    v4f o = {v, v, v, v};
    __builtin_nontemporal_store(o, &out[idx]);
}

// ---------------------------------------------------------------------------
extern "C" void kernel_launch(void* const* d_in, const int* in_sizes, int n_in,
                              void* d_out, int out_size, void* d_ws, size_t ws_size,
                              hipStream_t stream) {
    (void)in_sizes; (void)n_in; (void)out_size; (void)ws_size;
    const float* in_p  = (const float*)d_in[0];   // (32,4096,256)
    const float* A     = (const float*)d_in[1];   // (256,256)
    const float* Bv    = (const float*)d_in[2];   // (256,1)
    const float* Cv    = (const float*)d_in[3];   // (1,256)
    const float* delta = (const float*)d_in[4];   // (1,)
    const float* pw    = (const float*)d_in[5];   // (256,)
    const float* pb    = (const float*)d_in[6];   // (256,)

    char* ws = (char*)d_ws;
    size_t off = 0;
    auto take = [&](size_t bytes) { char* p = ws + off; off += (bytes + 255) & ~size_t(255); return p; };

    float*  flat   = (float*) take(131072 * 4);
    __bf16* flatbf = (__bf16*)take(131072 * 2);
    float*  dAf    = (float*) take(65536 * 4);
    __bf16* dAbf   = (__bf16*)take(65536 * 2);
    float*  Mf0    = (float*) take(65536 * 4);    // ping f32
    __bf16* Mb0    = (__bf16*)take(65536 * 2);    // ping bf16
    float*  Mf1    = (float*) take(65536 * 4);    // pong f32
    __bf16* Mb1    = (__bf16*)take(65536 * 2);    // pong bf16
    float*  dBv_   = (float*) take(256 * 4);
    float*  Smat   = (float*) take(64 * 256 * 4);
    float*  Vmat   = (float*) take(64 * 256 * 4);
    float*  kern   = (float*) take(4096 * 4);
    __bf16* gpad   = (__bf16*)take(8192 * 2);
    float*  res    = (float*) take(131072 * 4);

    // 1) dA = exp(delta*A), dB = delta*B
    ssm_prep_kernel<<<256, 256, 0, stream>>>(A, Bv, delta, dAf, dAbf, dBv_);

    // 2) flat = row-reduction of projected input (memory-bound read phase)
    ssm_proj_kernel<<<16384, 256, 0, stream>>>(in_p, pw, pb, flat, flatbf);

    // 3) P = dA^64 via 6 WMMA squarings (ping-pong)
    ssm_gemm256_kernel<<<32, 256, 0, stream>>>(dAbf, dAbf, Mf0, Mb0); // dA^2
    ssm_gemm256_kernel<<<32, 256, 0, stream>>>(Mb0,  Mb0,  Mf1, Mb1); // dA^4
    ssm_gemm256_kernel<<<32, 256, 0, stream>>>(Mb1,  Mb1,  Mf0, Mb0); // dA^8
    ssm_gemm256_kernel<<<32, 256, 0, stream>>>(Mb0,  Mb0,  Mf1, Mb1); // dA^16
    ssm_gemm256_kernel<<<32, 256, 0, stream>>>(Mb1,  Mb1,  Mf0, Mb0); // dA^32
    ssm_gemm256_kernel<<<32, 256, 0, stream>>>(Mb0,  Mb0,  Mf1, Mb1); // dA^64
    const __bf16* Pbf = Mb1;

    // 4) s_q / v_r chains: 64 steps each, LDS-staged bf16 matrices
    //    (128 KB matrix + 1 KB vec + 4 KB partials of dynamic LDS per block)
    ssm_chains_kernel<<<2, 1024, 136192, stream>>>(Pbf, dAbf, Cv, dBv_, Smat, Vmat);

    // 5) kernel[t] = s_q . v_r ; fold fwd+bwd into padded bf16 g
    ssm_outer_kernel<<<16, 256, 0, stream>>>(Smat, Vmat, kern);
    ssm_mkgpad_kernel<<<32, 256, 0, stream>>>(kern, gpad);

    // 6) causal Toeplitz GEMM via WMMA -> res(32 x 4096)
    ssm_conv_kernel<<<64, 256, 0, stream>>>(flatbf, gpad, res);

    // 7) broadcast over D (bandwidth-bound non-temporal write phase)
    ssm_bcast_kernel<<<32768, 256, 0, stream>>>(res, (v4f*)d_out);
}
